// MultiHeadAttention_9818295239040
// MI455X (gfx1250) — matrix operand order repaired and checked by reading, not compiled, not run on hardware
//
#include <hip/hip_runtime.h>
#include <math.h>

typedef __bf16 bf16_t;
typedef __bf16 v16bf __attribute__((ext_vector_type(16)));
typedef __bf16 v8bf  __attribute__((ext_vector_type(8)));
typedef __bf16 v4bf  __attribute__((ext_vector_type(4)));
typedef float  v8f   __attribute__((ext_vector_type(8)));

#define BM 128
#define BN 128
#define BK 32
#define LDT 48   // padded LDS tile stride (elements); 96B rows, 16B aligned

union Frag { v16bf v; v8bf h[2]; };

// A fragment (16x32, M x K): lane(lo) = row M; e=0..7 -> K=8*hi+e,
// e=8..15 -> K=16+8*hi+(e-8). Two contiguous 16B loads.
__device__ __forceinline__ v16bf load_frag_a(const bf16_t* rowptr, int lanehi, int kc) {
  Frag f;
  f.h[0] = *(const v8bf*)(rowptr + kc + 8 * lanehi);
  f.h[1] = *(const v8bf*)(rowptr + kc + 16 + 8 * lanehi);
  return f.v;
}

// B fragment (32x16, K x N) from an [N][K] layout: lane(lo) = col N;
// e=0..7 -> K=8*hi+e, e=8..15 -> K=16+8*hi+(e-8). Two contiguous 16B loads.
__device__ __forceinline__ v16bf load_frag_b(const bf16_t* colptr, int lanehi, int kc) {
  Frag f;
  f.h[0] = *(const v8bf*)(colptr + kc + 8 * lanehi);
  f.h[1] = *(const v8bf*)(colptr + kc + 16 + 8 * lanehi);
  return f.v;
}

__device__ __forceinline__ v8f wmma_bf16(v16bf a, v16bf b, v8f c) {
  return __builtin_amdgcn_wmma_f32_16x16x32_bf16(false, a, false, b, (short)0, c,
                                                 false, false);
}

// gfx1250 async global->LDS copy (16B per lane), tracked by ASYNCcnt.
// LDS address comes from the VDST operand; generic LDS ptr low 32 bits == LDS offset.
__device__ __forceinline__ void async_cp16(bf16_t* lds_dst, const bf16_t* gsrc) {
  const unsigned l = (unsigned)(uintptr_t)lds_dst;
  asm volatile("global_load_async_to_lds_b128 %0, %1, off"
               :: "v"(l), "v"(gsrc) : "memory");
}
__device__ __forceinline__ void wait_async0() {
  asm volatile("s_wait_asynccnt 0x0" ::: "memory");
}

// One 128x32 bf16 tile (1024-elem row stride in global) -> LDS, 2 async/thread.
__device__ __forceinline__ void load_tile_async(bf16_t* lds_base,
                                                const bf16_t* gbase, int tid) {
  const int row = tid >> 1;
  const int c0 = (tid & 1) * 16;
  #pragma unroll
  for (int j = 0; j < 2; ++j) {
    const int col = c0 + j * 8;
    async_cp16(lds_base + row * LDT + col, gbase + (size_t)row * 1024 + col);
  }
}

// ---------------------------------------------------------------------------
// Prep kernels: x -> bf16; W -> bf16 transposed [n][k].
// ---------------------------------------------------------------------------
__global__ __launch_bounds__(256)
void convert_x(const float* __restrict__ src, bf16_t* __restrict__ dst) {
  const size_t i = ((size_t)blockIdx.x * 256 + threadIdx.x) * 4;
  const float4 f = *(const float4*)(src + i);
  v4bf h;
  h[0] = (bf16_t)f.x; h[1] = (bf16_t)f.y; h[2] = (bf16_t)f.z; h[3] = (bf16_t)f.w;
  *(v4bf*)(dst + i) = h;
}

__global__ __launch_bounds__(256)
void transpose_w(const float* __restrict__ w0, const float* __restrict__ w1,
                 const float* __restrict__ w2, const float* __restrict__ w3,
                 bf16_t* __restrict__ wt) {
  __shared__ float tile[32][33];
  const int z = blockIdx.z;
  const float* w = (z == 0) ? w0 : (z == 1) ? w1 : (z == 2) ? w2 : w3;
  bf16_t* dst = wt + (size_t)z * 1024 * 1024;
  const int tx = threadIdx.x & 31, ty = threadIdx.x >> 5;
  const int n0 = blockIdx.x * 32, k0 = blockIdx.y * 32;
  #pragma unroll
  for (int j = 0; j < 4; ++j)
    tile[ty + j * 8][tx] = w[(size_t)(k0 + ty + j * 8) * 1024 + n0 + tx];
  __syncthreads();
  #pragma unroll
  for (int j = 0; j < 4; ++j)
    dst[(size_t)(n0 + ty + j * 8) * 1024 + k0 + tx] = (bf16_t)tile[tx][ty + j * 8];
}

// ---------------------------------------------------------------------------
// Kernel 1: Q/K/V = x @ W + b (all-bf16 tiles, async double-buffered LDS).
// Q,K stored [b,h,s,dh]; V stored transposed [b,h,dh,s].
// ---------------------------------------------------------------------------
__global__ __launch_bounds__(256)
void qkv_gemm(const bf16_t* __restrict__ xb,
              const bf16_t* __restrict__ wqt, const bf16_t* __restrict__ wkt,
              const bf16_t* __restrict__ wvt,
              const float* __restrict__ bq, const float* __restrict__ bk,
              const float* __restrict__ bv,
              bf16_t* __restrict__ q, bf16_t* __restrict__ k,
              bf16_t* __restrict__ v) {
  __shared__ bf16_t As[2][BM * LDT];
  __shared__ bf16_t Bs[2][BN * LDT];

  const int mat = blockIdx.z;
  const bf16_t* wt  = (mat == 0) ? wqt : (mat == 1) ? wkt : wvt;
  const float* bias = (mat == 0) ? bq  : (mat == 1) ? bk  : bv;
  bf16_t* outp      = (mat == 0) ? q   : (mat == 1) ? k   : v;

  const int tid = threadIdx.x;
  const int n0 = blockIdx.x * BN;
  const int m0 = blockIdx.y * BM;
  const int wave = tid >> 5, lane = tid & 31;
  const int lanelo = lane & 15, lanehi = lane >> 4;
  const int wm = (wave >> 1) * 32;  // wave tile: 32(M) x 64(N)
  const int wn = (wave & 1) * 64;

  const bf16_t* xa = xb + (size_t)m0 * 1024;
  const bf16_t* wb = wt + (size_t)n0 * 1024;

  v8f acc[2][4] = {};

  load_tile_async(&As[0][0], xa, tid);
  load_tile_async(&Bs[0][0], wb, tid);

  for (int it = 0; it < 32; ++it) {
    wait_async0();
    __syncthreads();
    const int cur = it & 1;
    if (it + 1 < 32) {
      load_tile_async(&As[cur ^ 1][0], xa + (it + 1) * BK, tid);
      load_tile_async(&Bs[cur ^ 1][0], wb + (it + 1) * BK, tid);
    }
    v16bf afr[2], bfr[4];
    #pragma unroll
    for (int mi = 0; mi < 2; ++mi)
      afr[mi] = load_frag_a(&As[cur][0] + (wm + mi * 16 + lanelo) * LDT, lanehi, 0);
    #pragma unroll
    for (int ni = 0; ni < 4; ++ni)
      bfr[ni] = load_frag_b(&Bs[cur][0] + (wn + ni * 16 + lanelo) * LDT, lanehi, 0);
    #pragma unroll
    for (int mi = 0; mi < 2; ++mi)
      #pragma unroll
      for (int ni = 0; ni < 4; ++ni)
        acc[mi][ni] = wmma_bf16(afr[mi], bfr[ni], acc[mi][ni]);
  }

  // Epilogue: +bias, split heads, bf16 store (V transposed).
  #pragma unroll
  for (int mi = 0; mi < 2; ++mi)
    #pragma unroll
    for (int ni = 0; ni < 4; ++ni) {
      const int n = n0 + wn + ni * 16 + lanelo;
      const int h = n >> 6, dh = n & 63;
      const float bv_ = bias[n];
      #pragma unroll
      for (int r = 0; r < 8; ++r) {
        const int m = m0 + wm + mi * 16 + r + 8 * lanehi;  // C/D: M = r + 8*hi
        const int b = m >> 6, s = m & 63;
        const float cval = acc[mi][ni][r] + bv_;
        size_t off;
        if (mat == 2) off = ((size_t)(b * 16 + h) * 64 + dh) * 64 + s;  // V^T
        else          off = ((size_t)(b * 16 + h) * 64 + s) * 64 + dh;
        outp[off] = (bf16_t)cval;
      }
    }
}

// ---------------------------------------------------------------------------
// Kernel 2: per-(b,h) attention. 4 waves x 16 query rows.
// ---------------------------------------------------------------------------
__global__ __launch_bounds__(128)
void attn_kernel(const bf16_t* __restrict__ q, const bf16_t* __restrict__ k,
                 const bf16_t* __restrict__ v, bf16_t* __restrict__ ctx) {
  __shared__ bf16_t attnS[64 * 72];

  const int bh = blockIdx.x;  // b*16 + h
  const int tid = threadIdx.x;
  const int wave = tid >> 5, lane = tid & 31;
  const int lanelo = lane & 15, lanehi = lane >> 4;
  const int srow0 = wave * 16;

  const bf16_t* qb = q + (size_t)bh * 4096;
  const bf16_t* kb = k + (size_t)bh * 4096;
  const bf16_t* vb = v + (size_t)bh * 4096;  // [dh][s] layout

  v8f acc[4] = {};
  #pragma unroll
  for (int kc = 0; kc < 64; kc += 32) {
    const v16bf af = load_frag_a(qb + (srow0 + lanelo) * 64, lanehi, kc);
    #pragma unroll
    for (int ni = 0; ni < 4; ++ni) {
      const v16bf bfv = load_frag_b(kb + (ni * 16 + lanelo) * 64, lanehi, kc);
      acc[ni] = wmma_bf16(af, bfv, acc[ni]);
    }
  }

  const float scale = 0.03125f;  // 1/sqrt(1024)
  float vals[4][8], rowmax[8], rowsum[8];
  #pragma unroll
  for (int r = 0; r < 8; ++r) rowmax[r] = -3.4028235e38f;
  #pragma unroll
  for (int ni = 0; ni < 4; ++ni)
    #pragma unroll
    for (int r = 0; r < 8; ++r) {
      float vv = acc[ni][r] * scale;
      if (vv == 0.0f) vv = 1e-9f;  // reference's (score==0)*1e-9 hack
      vals[ni][r] = vv;
      rowmax[r] = fmaxf(rowmax[r], vv);
    }
  #pragma unroll
  for (int r = 0; r < 8; ++r)
    for (int msk = 1; msk < 16; msk <<= 1)  // stays within 16-lane half
      rowmax[r] = fmaxf(rowmax[r], __shfl_xor(rowmax[r], msk, 32));
  #pragma unroll
  for (int r = 0; r < 8; ++r) rowsum[r] = 0.0f;
  #pragma unroll
  for (int ni = 0; ni < 4; ++ni)
    #pragma unroll
    for (int r = 0; r < 8; ++r) {
      const float e = __expf(vals[ni][r] - rowmax[r]);
      vals[ni][r] = e;
      rowsum[r] += e;
    }
  #pragma unroll
  for (int r = 0; r < 8; ++r)
    for (int msk = 1; msk < 16; msk <<= 1)
      rowsum[r] += __shfl_xor(rowsum[r], msk, 32);

  #pragma unroll
  for (int r = 0; r < 8; ++r) {
    const float inv = 1.0f / rowsum[r];
    #pragma unroll
    for (int ni = 0; ni < 4; ++ni)
      attnS[(srow0 + r + 8 * lanehi) * 72 + ni * 16 + lanelo] =
          (bf16_t)(vals[ni][r] * inv);
  }
  __syncthreads();

  v8f cacc[4] = {};
  #pragma unroll
  for (int kc = 0; kc < 64; kc += 32) {
    const v16bf af = load_frag_a(attnS + (srow0 + lanelo) * 72, lanehi, kc);
    #pragma unroll
    for (int ni = 0; ni < 4; ++ni) {
      const v16bf bfv = load_frag_b(vb + (ni * 16 + lanelo) * 64, lanehi, kc);
      cacc[ni] = wmma_bf16(af, bfv, cacc[ni]);
    }
  }

  const int b = bh >> 4, h = bh & 15;
  #pragma unroll
  for (int ni = 0; ni < 4; ++ni) {
    const int dh = ni * 16 + lanelo;
    #pragma unroll
    for (int r = 0; r < 8; ++r) {
      const int s = srow0 + r + 8 * lanehi;
      ctx[(size_t)(b * 64 + s) * 1024 + h * 64 + dh] = (bf16_t)cacc[ni][r];
    }
  }
}

// ---------------------------------------------------------------------------
// Kernel 3: out = gelu(ctx @ Wo + bo), exact erf gelu, fp32 output.
// ---------------------------------------------------------------------------
__global__ __launch_bounds__(256)
void out_gemm(const bf16_t* __restrict__ ctx, const bf16_t* __restrict__ wot,
              const float* __restrict__ bo, float* __restrict__ out) {
  __shared__ bf16_t As[2][BM * LDT];
  __shared__ bf16_t Bs[2][BN * LDT];

  const int tid = threadIdx.x;
  const int n0 = blockIdx.x * BN;
  const int m0 = blockIdx.y * BM;
  const int wave = tid >> 5, lane = tid & 31;
  const int lanelo = lane & 15, lanehi = lane >> 4;
  const int wm = (wave >> 1) * 32;
  const int wn = (wave & 1) * 64;

  const bf16_t* xa = ctx + (size_t)m0 * 1024;
  const bf16_t* wb = wot + (size_t)n0 * 1024;

  v8f acc[2][4] = {};

  load_tile_async(&As[0][0], xa, tid);
  load_tile_async(&Bs[0][0], wb, tid);

  for (int it = 0; it < 32; ++it) {
    wait_async0();
    __syncthreads();
    const int cur = it & 1;
    if (it + 1 < 32) {
      load_tile_async(&As[cur ^ 1][0], xa + (it + 1) * BK, tid);
      load_tile_async(&Bs[cur ^ 1][0], wb + (it + 1) * BK, tid);
    }
    v16bf afr[2], bfr[4];
    #pragma unroll
    for (int mi = 0; mi < 2; ++mi)
      afr[mi] = load_frag_a(&As[cur][0] + (wm + mi * 16 + lanelo) * LDT, lanehi, 0);
    #pragma unroll
    for (int ni = 0; ni < 4; ++ni)
      bfr[ni] = load_frag_b(&Bs[cur][0] + (wn + ni * 16 + lanelo) * LDT, lanehi, 0);
    #pragma unroll
    for (int mi = 0; mi < 2; ++mi)
      #pragma unroll
      for (int ni = 0; ni < 4; ++ni)
        acc[mi][ni] = wmma_bf16(afr[mi], bfr[ni], acc[mi][ni]);
  }

  #pragma unroll
  for (int mi = 0; mi < 2; ++mi)
    #pragma unroll
    for (int ni = 0; ni < 4; ++ni) {
      const int n = n0 + wn + ni * 16 + lanelo;
      const float bv_ = bo[n];
      #pragma unroll
      for (int r = 0; r < 8; ++r) {
        const int m = m0 + wm + mi * 16 + r + 8 * lanehi;
        const float cv = acc[mi][ni][r] + bv_;
        const float g = 0.5f * cv * (1.0f + erff(cv * 0.70710678118654752f));
        out[(size_t)m * 1024 + n] = g;
      }
    }
}

// ---------------------------------------------------------------------------
extern "C" void kernel_launch(void* const* d_in, const int* in_sizes, int n_in,
                              void* d_out, int out_size, void* d_ws, size_t ws_size,
                              hipStream_t stream) {
  const float* x    = (const float*)d_in[0];
  const float* wq_w = (const float*)d_in[1];
  const float* wq_b = (const float*)d_in[2];
  const float* wk_w = (const float*)d_in[3];
  const float* wk_b = (const float*)d_in[4];
  const float* wv_w = (const float*)d_in[5];
  const float* wv_b = (const float*)d_in[6];
  const float* wo_w = (const float*)d_in[7];
  const float* wo_b = (const float*)d_in[8];

  const size_t MK = (size_t)32768 * 1024;  // elements per activation tensor
  const size_t WK = (size_t)1024 * 1024;
  bf16_t* q   = (bf16_t*)d_ws;
  bf16_t* k   = q + MK;
  bf16_t* v   = k + MK;
  bf16_t* cx  = v + MK;
  bf16_t* xbf = cx + MK;
  bf16_t* wt  = xbf + MK;  // wq_t, wk_t, wv_t, wo_t back-to-back
  bf16_t* wqt = wt;
  bf16_t* wkt = wt + WK;
  bf16_t* wvt = wt + 2 * WK;
  bf16_t* wot = wt + 3 * WK;

  convert_x<<<dim3(32768), 256, 0, stream>>>(x, xbf);
  transpose_w<<<dim3(32, 32, 4), 256, 0, stream>>>(wq_w, wk_w, wv_w, wo_w, wt);
  qkv_gemm<<<dim3(1024 / BN, 32768 / BM, 3), 256, 0, stream>>>(
      xbf, wqt, wkt, wvt, wq_b, wk_b, wv_b, q, k, v);
  attn_kernel<<<dim3(512 * 16), 128, 0, stream>>>(q, k, v, cx);
  out_gemm<<<dim3(1024 / BN, 32768 / BM), 256, 0, stream>>>(cx, wot, wo_b,
                                                            (float*)d_out);
}
